// GCNLayers_74277164417597
// MI455X (gfx1250) — matrix operand order repaired and checked by reading, not compiled, not run on hardware
//
#include <hip/hip_runtime.h>

#define FDIM 256
#define INDIM 1280
#define NEG_SLOPE 0.01f
#define LN_EPS 1e-5f

typedef __attribute__((ext_vector_type(8)))  float  v8f;
typedef __attribute__((ext_vector_type(16))) __bf16 v16bf;
typedef __attribute__((ext_vector_type(2)))  __bf16 v2bf;

union FragU { unsigned u[8]; v16bf v; };

__device__ __forceinline__ unsigned short f2bf(float f) {
  unsigned u = __builtin_bit_cast(unsigned, f);
  u += 0x7fffu + ((u >> 16) & 1u);          // round to nearest even
  return (unsigned short)(u >> 16);
}

// pack two f32 -> bf16x2 in one uint (low = first arg)
__device__ __forceinline__ unsigned pack2bf(float lo, float hi) {
#if __has_builtin(__builtin_amdgcn_cvt_pk_bf16_f32)
  v2bf r = __builtin_amdgcn_cvt_pk_bf16_f32(lo, hi);
  return __builtin_bit_cast(unsigned, r);
#else
  return (unsigned)f2bf(lo) | ((unsigned)f2bf(hi) << 16);
#endif
}

__device__ __forceinline__ void waitAsync() {
#if __has_builtin(__builtin_amdgcn_s_wait_asynccnt)
  __builtin_amdgcn_s_wait_asynccnt(0);
#else
  asm volatile("s_wait_asynccnt 0x0" ::: "memory");
#endif
}

// -------------------------------------------------------------------------
// Weight pre-pass: W[K x 256] f32 (row-major) -> Wt[256 x K] bf16 (k-contig).
// -------------------------------------------------------------------------
__global__ __launch_bounds__(256)
void wt_convert(const float* __restrict__ W, unsigned short* __restrict__ Wt, int K) {
  int gid = blockIdx.x * 256 + threadIdx.x;
  if (gid >= K * FDIM) return;
  int k = gid >> 8, n = gid & 255;
  Wt[(size_t)n * K + k] = f2bf(W[gid]);
}

// -------------------------------------------------------------------------
// out[M x 256] = X[M x K] @ W[K x 256] (+ optional bias).
// X: f32, packed to bf16 at load (one uint4 of prefetch state per thread).
// Wt: pre-converted bf16 [256 x K]; B tiles stream via async global->LDS.
// Block: 256 thr = 8 waves (4 row-groups x 2 col-groups). Tile 64(M)x256(N),
// K-step 32, double-buffered LDS (40 KB of 320 KB WGP LDS).
// -------------------------------------------------------------------------
__global__ __launch_bounds__(256)
void gemm_bf16_wmma(const float* __restrict__ X, const unsigned short* __restrict__ Wt,
                    const float* __restrict__ bias, float* __restrict__ out,
                    int M, int K) {
  __shared__ unsigned short Ash[2][64 * 32];    // [m][k] bf16
  __shared__ unsigned short Bsh[2][256 * 32];   // [n][k] bf16

  const int tid  = threadIdx.x;
  const int wave = tid >> 5;
  const int lane = tid & 31;
  const int rg   = wave & 3;   // row group: rows rg*16..+15
  const int cg   = wave >> 2;  // col group: cols cg*128..+127
  const int m0   = blockIdx.x * 64;

  // A staging role: thread -> (row 0..63, koff 0/8/16/24), 8 floats each
  const int arow  = tid >> 2;
  const int akoff = (tid & 3) * 8;
  const int agrow = m0 + arow;
  // B staging role: thread t copies row n=t of Wt (32 bf16 = 64B per k-step)
  const unsigned long long wtRow =
      (unsigned long long)(uintptr_t)(Wt + (size_t)tid * K);

  v8f acc[8] = {};
  uint4 pa;

  auto loadA = [&](int k, uint4& pk) {
    if (agrow < M) {
      const float4* p = (const float4*)(X + (size_t)agrow * K + k + akoff);
      float4 a0 = p[0], a1 = p[1];
      pk.x = pack2bf(a0.x, a0.y);
      pk.y = pack2bf(a0.z, a0.w);
      pk.z = pack2bf(a1.x, a1.y);
      pk.w = pack2bf(a1.z, a1.w);
    } else {
      pk = make_uint4(0u, 0u, 0u, 0u);
    }
  };
  auto storeA = [&](int buf, const uint4& pk) {
    *(uint4*)&Ash[buf][arow * 32 + akoff] = pk;      // one ds_store_b128
  };
  auto asyncB = [&](int buf, int k) {
    // LDS generic pointer: low 32 bits are the LDS byte offset.
    unsigned lo = (unsigned)(uintptr_t)&Bsh[buf][tid * 32];
    unsigned long long ga = wtRow + (unsigned)(2 * k);
    // INST_OFFSET is added to both LDS and global addresses (ISA 08 §4.4).
    asm volatile(
        "global_load_async_to_lds_b128 %0, %1, off\n\t"
        "global_load_async_to_lds_b128 %0, %1, off offset:16\n\t"
        "global_load_async_to_lds_b128 %0, %1, off offset:32\n\t"
        "global_load_async_to_lds_b128 %0, %1, off offset:48"
        :: "v"(lo), "v"(ga) : "memory");
  };
  auto compute = [&](int buf) {
    const unsigned* A32 = (const unsigned*)Ash[buf];
    const unsigned* B32 = (const unsigned*)Bsh[buf];
    // A fragment (16x32 bf16): lane<16 -> K {0..15},
    // lane>=16 -> K {16..31}; uint j holds one K pair.
    FragU a;
    {
      int m   = rg * 16 + (lane & 15);
      int kh4 = (lane >> 4) * 8;                 // uint offset for K-half
      #pragma unroll
      for (int j = 0; j < 8; ++j) {
        int ku = j + kh4;                        // (kk>>1)
        a.u[j] = A32[m * 16 + ku];
      }
    }
    #pragma unroll
    for (int nt = 0; nt < 8; ++nt) {
      // B fragment (32x16): lanes 0-15 K 0-15, lanes 16-31 K 16-31.
      FragU b;
      int n  = cg * 128 + nt * 16 + (lane & 15);
      int kb = (lane >> 4) * 8;                  // uint offset
      #pragma unroll
      for (int j = 0; j < 8; ++j) b.u[j] = B32[n * 16 + j + kb];
      acc[nt] = __builtin_amdgcn_wmma_f32_16x16x32_bf16(
          false, a.v, false, b.v, (short)0, acc[nt], false, false);
    }
  };

  const int nk = K / 32;
  // prologue: fill buffer 0
  loadA(0, pa);
  asyncB(0, 0);
  storeA(0, pa);
  waitAsync();
  __syncthreads();

  int cur = 0;
  for (int kk = 0; kk + 1 < nk; ++kk) {      // steady state, no tail branch
    loadA((kk + 1) * 32, pa);                // global f32 -> packed bf16 regs
    asyncB(cur ^ 1, (kk + 1) * 32);          // async global->LDS for B
    compute(cur);                            // ds_read + 8x WMMA per wave
    storeA(cur ^ 1, pa);                     // one ds_store_b128
    waitAsync();
    __syncthreads();
    cur ^= 1;
  }
  compute(cur);                              // peeled last tile

  // epilogue: C/D layout — VGPR r: lanes<16 -> M=r, lanes>=16 -> M=r+8
  const int mlocal = rg * 16 + ((lane >> 4) ? 8 : 0);
  const int colb   = cg * 128 + (lane & 15);
  #pragma unroll
  for (int nt = 0; nt < 8; ++nt) {
    int col  = colb + nt * 16;
    float bv = bias ? bias[col] : 0.0f;
    #pragma unroll
    for (int r = 0; r < 8; ++r) {
      int row = m0 + mlocal + r;
      if (row < M) out[(size_t)row * FDIM + col] = acc[nt][r] + bv;
    }
  }
}

// -------------------------------------------------------------------------
// Row-wise LayerNorm + LeakyReLU. One wave32 per 256-wide row.
// -------------------------------------------------------------------------
__global__ __launch_bounds__(256)
void ln_leaky(const float* __restrict__ in, float* __restrict__ out,
              const float* __restrict__ gamma, const float* __restrict__ beta,
              int rows) {
  int row  = blockIdx.x * 8 + (threadIdx.x >> 5);
  int lane = threadIdx.x & 31;
  if (row >= rows) return;

  const float4* in4 = (const float4*)(in + (size_t)row * FDIM);
  float4 a = in4[lane * 2 + 0];
  float4 b = in4[lane * 2 + 1];

  float s = a.x + a.y + a.z + a.w + b.x + b.y + b.z + b.w;
  float q = a.x*a.x + a.y*a.y + a.z*a.z + a.w*a.w
          + b.x*b.x + b.y*b.y + b.z*b.z + b.w*b.w;
  #pragma unroll
  for (int off = 16; off > 0; off >>= 1) {
    s += __shfl_xor(s, off, 32);
    q += __shfl_xor(q, off, 32);
  }
  float mean = s * (1.0f / FDIM);
  float var  = q * (1.0f / FDIM) - mean * mean;
  float rstd = rsqrtf(var + LN_EPS);

  float vals[8] = {a.x, a.y, a.z, a.w, b.x, b.y, b.z, b.w};
  int c = lane * 8;
  float o[8];
  #pragma unroll
  for (int j = 0; j < 8; ++j) {
    float y = (vals[j] - mean) * rstd * gamma[c + j] + beta[c + j];
    o[j] = (y >= 0.0f) ? y : NEG_SLOPE * y;
  }
  float4* out4 = (float4*)(out + (size_t)row * FDIM);
  out4[lane * 2 + 0] = make_float4(o[0], o[1], o[2], o[3]);
  out4[lane * 2 + 1] = make_float4(o[4], o[5], o[6], o[7]);
}

// -------------------------------------------------------------------------
// Degree / norm helpers
// -------------------------------------------------------------------------
__global__ void deg_init(float* __restrict__ deg, int n) {
  int i = blockIdx.x * 256 + threadIdx.x;
  if (i < n) deg[i] = 1.0f;  // self-loop
}
__global__ void deg_count(const int* __restrict__ dst, int e, float* __restrict__ deg) {
  int i = blockIdx.x * 256 + threadIdx.x;
  if (i < e) atomicAdd(&deg[dst[i]], 1.0f);
}
__global__ void deg_rsqrt(float* __restrict__ deg, int n) {
  int i = blockIdx.x * 256 + threadIdx.x;
  if (i < n) deg[i] = rsqrtf(deg[i]);
}

// agg[v] = bias + hw[v] * dinv[v]^2   (bias + self-loop term)
__global__ __launch_bounds__(256)
void agg_init(const float* __restrict__ hw, const float* __restrict__ dinv,
              const float* __restrict__ bias, float* __restrict__ agg, int n) {
  int gid = blockIdx.x * 256 + threadIdx.x;
  if (gid >= n * 64) return;
  int node = gid >> 6, c = gid & 63;
  float w  = dinv[node];
  float w2 = w * w;
  float4 v  = ((const float4*)hw)[gid];
  float4 bb = ((const float4*)bias)[c];
  float4 o  = make_float4(bb.x + v.x * w2, bb.y + v.y * w2,
                          bb.z + v.z * w2, bb.w + v.w * w2);
  ((float4*)agg)[gid] = o;
}

// agg[dst[e]] += hw[src[e]] * dinv[src]*dinv[dst]; 64 lanes/edge, float4/lane.
// Gather source and scatter target are both L2-resident (51 MB each, 192 MB L2).
__global__ __launch_bounds__(256)
void edge_scatter(const float* __restrict__ hw, const int* __restrict__ src,
                  const int* __restrict__ dst, const float* __restrict__ dinv,
                  float* __restrict__ agg, int e) {
  int gid = blockIdx.x * 256 + threadIdx.x;
  int ed  = gid >> 6;
  if (ed >= e) return;
  int c = gid & 63;
  int s = src[ed], d = dst[ed];
  float nrm = dinv[s] * dinv[d];
  float4 v = ((const float4*)hw)[(size_t)s * 64 + c];
  float* p = agg + (size_t)d * FDIM + c * 4;
  atomicAdd(p + 0, v.x * nrm);
  atomicAdd(p + 1, v.y * nrm);
  atomicAdd(p + 2, v.z * nrm);
  atomicAdd(p + 3, v.w * nrm);
}

// -------------------------------------------------------------------------
extern "C" void kernel_launch(void* const* d_in, const int* in_sizes, int n_in,
                              void* d_out, int out_size, void* d_ws, size_t ws_size,
                              hipStream_t stream) {
  (void)n_in; (void)out_size; (void)ws_size;
  const float* x[2]   = {(const float*)d_in[0], (const float*)d_in[2]};
  const int*   ei[2]  = {(const int*)d_in[1],   (const int*)d_in[3]};
  const float* w_lin  = (const float*)d_in[4];
  const float* b_lin  = (const float*)d_in[5];
  const float* g_lin  = (const float*)d_in[6];
  const float* be_lin = (const float*)d_in[7];
  const float* w_gcn  = (const float*)d_in[8];   // [2,256,256]
  const float* b_gcn  = (const float*)d_in[9];   // [2,256]
  const float* g_act  = (const float*)d_in[10];
  const float* be_act = (const float*)d_in[11];

  const int N = in_sizes[0] / INDIM;
  const int E = in_sizes[1] / 2;
  float* out = (float*)d_out;

  // workspace layout
  char* ws = (char*)d_ws;
  size_t off = 0;
  auto take = [&](size_t bytes) {
    void* p = ws + off;
    off += (bytes + 255) & ~(size_t)255;
    return p;
  };
  size_t featBytes = (size_t)N * FDIM * sizeof(float);
  float* ping = (float*)take(featBytes);
  float* pong = (float*)take(featBytes);
  float* dinv = (float*)take((size_t)N * sizeof(float));
  unsigned short* wtLin = (unsigned short*)take((size_t)INDIM * FDIM * 2);
  unsigned short* wtG   = (unsigned short*)take((size_t)2 * FDIM * FDIM * 2);

  dim3 gemmGrid((unsigned)((N + 63) / 64));
  int lnBlocks   = (N + 7) / 8;
  int nodeBlocks = (N + 255) / 256;
  int edgeBlocks = (E + 255) / 256;
  int aggBlocks  = (N * 64 + 255) / 256;
  int scatBlocks = (E * 64 + 255) / 256;

  // one-time (per call) weight convert+transpose to bf16 [n][k]
  wt_convert<<<INDIM, 256, 0, stream>>>(w_lin, wtLin, INDIM);
  wt_convert<<<FDIM, 256, 0, stream>>>(w_gcn, wtG, FDIM);
  wt_convert<<<FDIM, 256, 0, stream>>>(w_gcn + (size_t)FDIM * FDIM,
                                       wtG + (size_t)FDIM * FDIM, FDIM);

  for (int g = 0; g < 2; ++g) {
    const int* src = ei[g];
    const int* dst = ei[g] + E;

    // symmetric-norm degree vector (constant across layers)
    deg_init <<<nodeBlocks, 256, 0, stream>>>(dinv, N);
    deg_count<<<edgeBlocks, 256, 0, stream>>>(dst, E, dinv);
    deg_rsqrt<<<nodeBlocks, 256, 0, stream>>>(dinv, N);

    // input projection + LN + LeakyReLU
    gemm_bf16_wmma<<<gemmGrid, 256, 0, stream>>>(x[g], wtLin, b_lin, pong, N, INDIM);
    ln_leaky<<<lnBlocks, 256, 0, stream>>>(pong, ping, g_lin, be_lin, N);

    for (int l = 0; l < 2; ++l) {
      gemm_bf16_wmma<<<gemmGrid, 256, 0, stream>>>(
          ping, wtG + (size_t)l * FDIM * FDIM, nullptr, pong, N, FDIM);
      agg_init<<<aggBlocks, 256, 0, stream>>>(pong, dinv, b_gcn + (size_t)l * FDIM, ping, N);
      edge_scatter<<<scatBlocks, 256, 0, stream>>>(pong, src, dst, dinv, ping, E);
      float* lnout = (l == 1) ? out + (size_t)g * N * FDIM : ping;
      ln_leaky<<<lnBlocks, 256, 0, stream>>>(ping, lnout,
                                             g_act + (size_t)l * FDIM,
                                             be_act + (size_t)l * FDIM, N);
    }
  }
}
